// MultiHeadSelfAttention_910533066929
// MI455X (gfx1250) — compile-verified, matrix operand order checked by reading, not run on hardware
//
#include <hip/hip_runtime.h>

// ---------------------------------------------------------------------------
// MI455X (gfx1250) multi-head self-attention, bf16 WMMA + f32 accumulate.
// B=2, S=2048, D=1024, H=16, dk=64.
// Compute-bound (~69 GFLOP vs ~50MB traffic): everything matrix-shaped runs
// on v_wmma_f32_16x16x32_bf16; K/V tiles staged to LDS with the gfx1250
// async-to-LDS path (ASYNCcnt) when the toolchain exposes it.
// ---------------------------------------------------------------------------

constexpr int Bn   = 2;
constexpr int Sn   = 2048;
constexpr int Dn   = 1024;
constexpr int Hn   = 16;
constexpr int DKn  = 64;
constexpr int BSn  = Bn * Sn;          // 4096 rows for the big GEMMs

typedef __attribute__((ext_vector_type(16))) __bf16 v16bf;
typedef __attribute__((ext_vector_type(8)))  __bf16 bf16x8;
typedef __attribute__((ext_vector_type(8)))  float  v8f;

#if defined(__AMDGCN__) && __has_builtin(__builtin_amdgcn_global_load_async_to_lds_b128)
#define HAVE_ASYNC_LDS 1
#endif

#if defined(HAVE_ASYNC_LDS)
// The builtin's pointer params are int4* in addrspace(1)/(3) (clang prints
// these as "int __device__ *" / "int __shared__ *" with vector_size(16)).
typedef int v4i_gnu __attribute__((vector_size(16)));
typedef __attribute__((address_space(1))) void*    as1_void_p;
typedef __attribute__((address_space(3))) void*    as3_void_p;
typedef __attribute__((address_space(1))) v4i_gnu* as1_v4i_p;
typedef __attribute__((address_space(3))) v4i_gnu* as3_v4i_p;
#endif

// ---------------------------------------------------------------------------
// WMMA helpers.
// ISA 16-bit A/B fragment layout (16x32, K contiguous in two 8-element runs):
//   element e of the per-lane v16bf maps to K(e) = (e<8) ? 8*half + e
//                                                        : 16 + 8*half + (e-8)
// so a fragment is two 16-byte contiguous loads per lane.
// ---------------------------------------------------------------------------
__device__ __forceinline__ v16bf load_frag_row(const __bf16* rowp, int half) {
    union { v16bf v; bf16x8 h[2]; } u;
    u.h[0] = *(const bf16x8*)(rowp + 8 * half);
    u.h[1] = *(const bf16x8*)(rowp + 16 + 8 * half);
    return u.v;
}

__device__ __forceinline__ v8f wmma_bf16(v16bf a, v16bf b, v8f c) {
    // (neg_a, A, neg_b, B, c_mod, C, reuse_a, reuse_b)
    return __builtin_amdgcn_wmma_f32_16x16x32_bf16(false, a, false, b,
                                                   (short)0, c, false, false);
}

// 16-byte global -> LDS copy, async (ASYNCcnt) when available.
__device__ __forceinline__ void copy_b128_to_lds(const void* g, void* l) {
#if defined(HAVE_ASYNC_LDS)
    __builtin_amdgcn_global_load_async_to_lds_b128(
        (as1_v4i_p)(as1_void_p)g,
        (as3_v4i_p)(as3_void_p)l, 0, 0);
#else
    *(bf16x8*)l = *(const bf16x8*)g;
#endif
}

__device__ __forceinline__ void wait_async_copies() {
#if defined(HAVE_ASYNC_LDS)
#if __has_builtin(__builtin_amdgcn_s_wait_asynccnt)
    __builtin_amdgcn_s_wait_asynccnt(0);
#else
    asm volatile("s_wait_asynccnt 0" ::: "memory");
#endif
#endif
}

#define V8F_ZERO {0.f,0.f,0.f,0.f,0.f,0.f,0.f,0.f}

// ---------------------------------------------------------------------------
// f32 -> bf16 elementwise convert
// ---------------------------------------------------------------------------
__global__ void f32_to_bf16_kernel(const float* __restrict__ in,
                                   __bf16* __restrict__ out, int n) {
    int i = blockIdx.x * blockDim.x + threadIdx.x;
    if (i < n) out[i] = (__bf16)in[i];
}

// ---------------------------------------------------------------------------
// NT GEMM: C[M][N] = sum_k A[m][k] * W[n][k], A:(M x K) bf16 row-major,
// W:(N x K) bf16 row-major, C f32 row-major. 256 threads = 8 waves;
// each wave computes a 32x64 tile (2 A-frags x 4 B-frags -> 8 WMMAs per
// 6 fragment loads); block covers 256(M) x 64(N). EXEC all-ones throughout.
// ---------------------------------------------------------------------------
__global__ __launch_bounds__(256)
void gemm_nt_bf16_kernel(const __bf16* __restrict__ A,
                         const __bf16* __restrict__ W,
                         float* __restrict__ C,
                         int M, int N, int K) {
    const int lane = threadIdx.x & 31;
    const int wave = threadIdx.x >> 5;
    const int half = lane >> 4;
    const int ln   = lane & 15;

    const int mbase = blockIdx.y * 256 + wave * 32;
    const int nbase = blockIdx.x * 64;

    v8f c00 = V8F_ZERO, c01 = V8F_ZERO, c02 = V8F_ZERO, c03 = V8F_ZERO;
    v8f c10 = V8F_ZERO, c11 = V8F_ZERO, c12 = V8F_ZERO, c13 = V8F_ZERO;

    const __bf16* a0r = A + (size_t)(mbase +  0 + ln) * K;
    const __bf16* a1r = A + (size_t)(mbase + 16 + ln) * K;
    const __bf16* b0r = W + (size_t)(nbase +  0 + ln) * K;
    const __bf16* b1r = W + (size_t)(nbase + 16 + ln) * K;
    const __bf16* b2r = W + (size_t)(nbase + 32 + ln) * K;
    const __bf16* b3r = W + (size_t)(nbase + 48 + ln) * K;

    for (int kk = 0; kk < K; kk += 32) {
        v16bf a0 = load_frag_row(a0r + kk, half);
        v16bf a1 = load_frag_row(a1r + kk, half);
        v16bf b0 = load_frag_row(b0r + kk, half);
        v16bf b1 = load_frag_row(b1r + kk, half);
        v16bf b2 = load_frag_row(b2r + kk, half);
        v16bf b3 = load_frag_row(b3r + kk, half);
        c00 = wmma_bf16(a0, b0, c00);
        c01 = wmma_bf16(a0, b1, c01);
        c02 = wmma_bf16(a0, b2, c02);
        c03 = wmma_bf16(a0, b3, c03);
        c10 = wmma_bf16(a1, b0, c10);
        c11 = wmma_bf16(a1, b1, c11);
        c12 = wmma_bf16(a1, b2, c12);
        c13 = wmma_bf16(a1, b3, c13);
    }

    // C/D layout: element r -> row (r + 8*half), col = ln within 16-col tile
#pragma unroll
    for (int r = 0; r < 8; ++r) {
        int m0 = mbase + r + 8 * half;
        float* crow0 = C + (size_t)m0 * N + nbase + ln;
        crow0[ 0] = c00[r];
        crow0[16] = c01[r];
        crow0[32] = c02[r];
        crow0[48] = c03[r];
        float* crow1 = crow0 + (size_t)16 * N;
        crow1[ 0] = c10[r];
        crow1[16] = c11[r];
        crow1[32] = c12[r];
        crow1[48] = c13[r];
    }
}

// ---------------------------------------------------------------------------
// RoPE + repack: f32 [B,S,H*dk] -> bf16 [B,H,S,dk], one thread per (even,odd)
// pair.  inv_freq = theta^(-2p/dk), angle = pos[s]*inv_freq.
// ---------------------------------------------------------------------------
__global__ void rope_pack_kernel(const float* __restrict__ Xf,
                                 const int* __restrict__ pos,
                                 __bf16* __restrict__ Out) {
    int idx = blockIdx.x * blockDim.x + threadIdx.x;
    const int total = Bn * Sn * Hn * (DKn / 2);
    if (idx >= total) return;
    int p = idx % (DKn / 2);
    int h = (idx / (DKn / 2)) % Hn;
    int s = (idx / (DKn / 2 * Hn)) % Sn;
    int b = idx / (DKn / 2 * Hn * Sn);

    size_t src = ((size_t)(b * Sn + s)) * Dn + h * DKn + 2 * p;
    float e = Xf[src];
    float o = Xf[src + 1];

    float inv = __powf(10000.0f, -((float)(2 * p) / (float)DKn));
    float ang = (float)pos[s] * inv;
    float sn, cs;
    __sincosf(ang, &sn, &cs);
    float re = e * cs - o * sn;
    float ro = o * cs + e * sn;

    size_t dst = (((size_t)(b * Hn + h)) * Sn + s) * DKn + 2 * p;
    Out[dst]     = (__bf16)re;
    Out[dst + 1] = (__bf16)ro;
}

// ---------------------------------------------------------------------------
// V repack transposed: f32 [B,S,H*dk] -> bf16 [B,H,dk,S] so P.V is NT.
// ---------------------------------------------------------------------------
__global__ void v_pack_t_kernel(const float* __restrict__ Xf,
                                __bf16* __restrict__ Vt) {
    int idx = blockIdx.x * blockDim.x + threadIdx.x;
    const int total = Bn * Sn * Hn * DKn;
    if (idx >= total) return;
    int d = idx % DKn;
    int h = (idx / DKn) % Hn;
    int s = (idx / (DKn * Hn)) % Sn;
    int b = idx / (DKn * Hn * Sn);
    float v = Xf[((size_t)(b * Sn + s)) * Dn + h * DKn + d];
    Vt[(((size_t)(b * Hn + h)) * DKn + d) * Sn + s] = (__bf16)v;
}

// ---------------------------------------------------------------------------
// Flash attention (causal, online softmax). 128 threads = 4 waves; wave w
// owns q-rows [qb + 16w, +16). Per 32-wide K chunk the whole workgroup
// cooperatively stages the K tile (32x64) and V tile (64x32, from Vt) into
// LDS via async-to-LDS, then every wave does 4 score WMMAs, online softmax
// (width-16 shuffles match the C-tile layout M=r+8*half / N=lane&15),
// bounces P through per-wave LDS to re-form an A fragment, and 4 P.V WMMAs.
// All waves iterate to the workgroup causal limit so barriers stay uniform;
// chunks above a wave's own rows are fully masked (exp -> 0, exact no-op).
// ---------------------------------------------------------------------------
__global__ __launch_bounds__(128)
void flash_attn_kernel(const __bf16* __restrict__ Qh,   // [B,H,S,dk]
                       const __bf16* __restrict__ Kh,   // [B,H,S,dk]
                       const __bf16* __restrict__ Vt,   // [B,H,dk,S]
                       __bf16* __restrict__ attn) {     // [B,S,H*dk]
    __shared__ __bf16 Kt[32][64];      // rows kc..kc+31, cols d      (4 KB)
    __shared__ __bf16 Vs[64][32];      // rows d, cols kc..kc+31      (4 KB)
    __shared__ __bf16 plds[4][16][32]; // per-wave P bounce           (4 KB)

    const int lane = threadIdx.x & 31;
    const int wave = threadIdx.x >> 5;
    const int half = lane >> 4;
    const int ln   = lane & 15;

    const int bh    = blockIdx.y;               // b*H + h
    const int qb    = blockIdx.x * 64;
    const int qbase = qb + wave * 16;

    const __bf16* Qp = Qh + (size_t)bh * Sn * DKn;
    const __bf16* Kp = Kh + (size_t)bh * Sn * DKn;
    const __bf16* Vp = Vt + (size_t)bh * DKn * Sn;

    // Q fragments for this wave's 16 rows (row = qbase + ln), dk = 2 x K32
    v16bf qa0 = load_frag_row(Qp + (size_t)(qbase + ln) * DKn +  0, half);
    v16bf qa1 = load_frag_row(Qp + (size_t)(qbase + ln) * DKn + 32, half);

    v8f o0 = V8F_ZERO, o1 = V8F_ZERO, o2 = V8F_ZERO, o3 = V8F_ZERO;
    float rmax[8], rsum[8];
#pragma unroll
    for (int r = 0; r < 8; ++r) { rmax[r] = -3.0e38f; rsum[r] = 0.0f; }

    const float scale = 0.125f;                 // 1/sqrt(64)
    const int   kend_wg = qb + 64;              // workgroup causal limit

    for (int kc = 0; kc < kend_wg; kc += 32) {
        __syncthreads();   // previous tiles fully consumed before overwrite

        // ---- cooperative async staging: K 4KB + V 4KB = 512 x b128 ----
        const int t = threadIdx.x;
#pragma unroll
        for (int i = 0; i < 2; ++i) {
            int li = t + 128 * i;                 // 0..255
            int kr = li >> 3, kcol = (li & 7) * 8;   // 32 rows x 64 elems
            copy_b128_to_lds(Kp + (size_t)(kc + kr) * DKn + kcol,
                             &Kt[kr][kcol]);
            int vr = li >> 2, vcol = (li & 3) * 8;   // 64 rows x 32 elems
            copy_b128_to_lds(Vp + (size_t)vr * Sn + kc + vcol,
                             &Vs[vr][vcol]);
        }
        wait_async_copies();
        __syncthreads();

        // ---- scores: S = Q . K^T, fragments from LDS ----
        v8f sc1 = V8F_ZERO, sc2 = V8F_ZERO;
        {
            v16bf kb0 = load_frag_row(&Kt[ln][0],       half);
            v16bf kb1 = load_frag_row(&Kt[ln][32],      half);
            v16bf kb2 = load_frag_row(&Kt[16 + ln][0],  half);
            v16bf kb3 = load_frag_row(&Kt[16 + ln][32], half);
            sc1 = wmma_bf16(qa0, kb0, sc1);
            sc1 = wmma_bf16(qa1, kb1, sc1);
            sc2 = wmma_bf16(qa0, kb2, sc2);
            sc2 = wmma_bf16(qa1, kb3, sc2);
        }

        // ---- online softmax (row m = r + 8*half, col = ln within tile) ----
#pragma unroll
        for (int r = 0; r < 8; ++r) {
            const int qrow = qbase + r + 8 * half;
            float s1 = sc1[r] * scale;
            if (kc + ln > qrow) s1 = -1.0e9f;
            float s2 = sc2[r] * scale;
            if (kc + 16 + ln > qrow) s2 = -1.0e9f;

            float mx = fmaxf(s1, s2);
            mx = fmaxf(mx, __shfl_xor(mx, 1, 16));
            mx = fmaxf(mx, __shfl_xor(mx, 2, 16));
            mx = fmaxf(mx, __shfl_xor(mx, 4, 16));
            mx = fmaxf(mx, __shfl_xor(mx, 8, 16));

            float nmax = fmaxf(rmax[r], mx);
            float corr = __expf(rmax[r] - nmax);
            float p1 = __expf(s1 - nmax);
            float p2 = __expf(s2 - nmax);

            float ps = p1 + p2;
            ps += __shfl_xor(ps, 1, 16);
            ps += __shfl_xor(ps, 2, 16);
            ps += __shfl_xor(ps, 4, 16);
            ps += __shfl_xor(ps, 8, 16);

            rsum[r] = rsum[r] * corr + ps;
            rmax[r] = nmax;
            o0[r] *= corr; o1[r] *= corr; o2[r] *= corr; o3[r] *= corr;

            plds[wave][r + 8 * half][ln]      = (__bf16)p1;
            plds[wave][r + 8 * half][16 + ln] = (__bf16)p2;
        }

        // per-wave LDS RAW: DS ops are in-order per wave; fence for safety
        asm volatile("s_wait_dscnt 0" ::: "memory");

        // ---- O += P . V  (NT against Vs[d][s-chunk], LDS) ----
        v16bf pa = load_frag_row(&plds[wave][ln][0], half);
        {
            v16bf vb0 = load_frag_row(&Vs[ 0 + ln][0], half);
            v16bf vb1 = load_frag_row(&Vs[16 + ln][0], half);
            v16bf vb2 = load_frag_row(&Vs[32 + ln][0], half);
            v16bf vb3 = load_frag_row(&Vs[48 + ln][0], half);
            o0 = wmma_bf16(pa, vb0, o0);
            o1 = wmma_bf16(pa, vb1, o1);
            o2 = wmma_bf16(pa, vb2, o2);
            o3 = wmma_bf16(pa, vb3, o3);
        }
    }

    // ---- normalize and store attn [B,S,H*dk] as bf16 ----
    const int b = bh >> 4;   // / Hn
    const int h = bh & 15;   // % Hn
#pragma unroll
    for (int r = 0; r < 8; ++r) {
        const int srow = qbase + r + 8 * half;
        const float inv = 1.0f / rsum[r];
        __bf16* arow = attn + ((size_t)(b * Sn + srow)) * Dn + h * DKn + ln;
        arow[ 0] = (__bf16)(o0[r] * inv);
        arow[16] = (__bf16)(o1[r] * inv);
        arow[32] = (__bf16)(o2[r] * inv);
        arow[48] = (__bf16)(o3[r] * inv);
    }
}

// ---------------------------------------------------------------------------
// Host launcher
// ---------------------------------------------------------------------------
extern "C" void kernel_launch(void* const* d_in, const int* in_sizes, int n_in,
                              void* d_out, int out_size, void* d_ws, size_t ws_size,
                              hipStream_t stream) {
    const float* x   = (const float*)d_in[0];
    const int*   pos = (const int*)  d_in[1];
    const float* Wq  = (const float*)d_in[2];
    const float* Wk  = (const float*)d_in[3];
    const float* Wv  = (const float*)d_in[4];
    const float* Wo  = (const float*)d_in[5];

    char* ws = (char*)d_ws;
    size_t off = 0;
    auto carve = [&](size_t bytes) -> void* {
        void* p = ws + off;
        off += (bytes + 255) & ~(size_t)255;
        return p;
    };

    __bf16* xb   = (__bf16*)carve((size_t)BSn * Dn * 2);
    __bf16* wqb  = (__bf16*)carve((size_t)Dn * Dn * 2);
    __bf16* wkb  = (__bf16*)carve((size_t)Dn * Dn * 2);
    __bf16* wvb  = (__bf16*)carve((size_t)Dn * Dn * 2);
    __bf16* wob  = (__bf16*)carve((size_t)Dn * Dn * 2);
    float*  scr  = (float*) carve((size_t)BSn * Dn * 4);
    __bf16* Qh   = (__bf16*)carve((size_t)BSn * Dn * 2);
    __bf16* Kh   = (__bf16*)carve((size_t)BSn * Dn * 2);
    __bf16* Vt   = (__bf16*)carve((size_t)BSn * Dn * 2);
    __bf16* attn = (__bf16*)carve((size_t)BSn * Dn * 2);

    const int nX = BSn * Dn;     // 4,194,304
    const int nW = Dn * Dn;      // 1,048,576
    f32_to_bf16_kernel<<<(nX + 255) / 256, 256, 0, stream>>>(x,  xb,  nX);
    f32_to_bf16_kernel<<<(nW + 255) / 256, 256, 0, stream>>>(Wq, wqb, nW);
    f32_to_bf16_kernel<<<(nW + 255) / 256, 256, 0, stream>>>(Wk, wkb, nW);
    f32_to_bf16_kernel<<<(nW + 255) / 256, 256, 0, stream>>>(Wv, wvb, nW);
    f32_to_bf16_kernel<<<(nW + 255) / 256, 256, 0, stream>>>(Wo, wob, nW);

    dim3 ggrid(Dn / 64, BSn / 256);    // (16, 16)
    const int nPair = Bn * Sn * Hn * (DKn / 2);

    // Q
    gemm_nt_bf16_kernel<<<ggrid, 256, 0, stream>>>(xb, wqb, scr, BSn, Dn, Dn);
    rope_pack_kernel<<<(nPair + 255) / 256, 256, 0, stream>>>(scr, pos, Qh);
    // K
    gemm_nt_bf16_kernel<<<ggrid, 256, 0, stream>>>(xb, wkb, scr, BSn, Dn, Dn);
    rope_pack_kernel<<<(nPair + 255) / 256, 256, 0, stream>>>(scr, pos, Kh);
    // V
    gemm_nt_bf16_kernel<<<ggrid, 256, 0, stream>>>(xb, wvb, scr, BSn, Dn, Dn);
    v_pack_t_kernel<<<(nX + 255) / 256, 256, 0, stream>>>(scr, Vt);

    // attention
    dim3 fgrid(Sn / 64, Bn * Hn);      // (32, 32)
    flash_attn_kernel<<<fgrid, 128, 0, stream>>>(Qh, Kh, Vt, attn);

    // out = attn @ Wo^T  -> f32 d_out
    gemm_nt_bf16_kernel<<<ggrid, 256, 0, stream>>>(attn, wob, (float*)d_out,
                                                   BSn, Dn, Dn);
}
